// GAT_26018911879763
// MI455X (gfx1250) — matrix operand order repaired and checked by reading, not compiled, not run on hardware
//
#include <hip/hip_runtime.h>
#include <hip/hip_bf16.h>

// ---------------------------------------------------------------------------
// GATv2 x3 + mean-pool for MI455X (gfx1250).
// GEMMs (node linears + fused edge-attr linear) run on v_wmma_f32_16x16x32_bf16
// (wave32, 16x16 tiles, f32 accumulate). Edge-attr GEMM is fused into the
// attention-score kernel so the 550000 x 256 "ee" tensor is never stored.
// Scatter kernels are float4-vectorized: the workload is HBM-bound (23.3 TB/s),
// so per-edge metadata loads are amortized over 4 channels.
// ---------------------------------------------------------------------------

typedef __bf16 bf16_t;
typedef __bf16 v16bf __attribute__((ext_vector_type(16)));
typedef __bf16 v8bf  __attribute__((ext_vector_type(8)));
typedef float  v8f   __attribute__((ext_vector_type(8)));

#define NNODES 50000
#define NEDGES 500000
#define NE2    550000   // edges + self loops
#define NGRAPH 50

__device__ __forceinline__ v8f wmma_bf16(v16bf a, v16bf b, v8f c) {
  // (neg_a, A, neg_b, B, c_mod, C, reuse_a, reuse_b)
  return __builtin_amdgcn_wmma_f32_16x16x32_bf16(false, a, false, b, (short)0, c,
                                                 false, false);
}

// A-matrix (16x32 bf16) fragment: lanes 0-15 are rows M=0..15 holding K
// [0..15]; lanes 16-31 same rows, K [16..31] (matches contiguous B loads).
__device__ __forceinline__ v16bf load_a_frag(const bf16_t* row, int g, int kbase) {
  v8bf lo = *(const v8bf*)(row + kbase + g * 16);
  v8bf hi = *(const v8bf*)(row + kbase + g * 16 + 8);
  v16bf a;
#pragma unroll
  for (int i = 0; i < 8; ++i) { a[i] = lo[i]; a[i + 8] = hi[i]; }
  return a;
}

// Monotone float<->uint mapping for atomic max on f32.
__device__ __forceinline__ unsigned fmap(float f) {
  unsigned u = __float_as_uint(f);
  return (u & 0x80000000u) ? ~u : (u | 0x80000000u);
}
__device__ __forceinline__ float funmap(unsigned m) {
  unsigned u = (m & 0x80000000u) ? (m ^ 0x80000000u) : ~m;
  return __uint_as_float(u);
}

__device__ __forceinline__ int src_of(const int* eidx, int e) {
  return (e < NEDGES) ? eidx[e] : (e - NEDGES);
}
__device__ __forceinline__ int dst_of(const int* eidx, int e) {
  return (e < NEDGES) ? eidx[NEDGES + e] : (e - NEDGES);
}

// ----------------------------- elementwise ---------------------------------
__global__ void k_zero_f32(float* p, long n) {
  long i = (long)blockIdx.x * blockDim.x + threadIdx.x;
  if (i < n) p[i] = 0.0f;
}
__global__ void k_init_u32(unsigned* p, long n, unsigned v) {
  long i = (long)blockIdx.x * blockDim.x + threadIdx.x;
  if (i < n) p[i] = v;
}
__global__ void k_f2bf(bf16_t* dst, const float* src, long n) {
  long i = (long)blockIdx.x * blockDim.x + threadIdx.x;
  if (i < n) dst[i] = (bf16_t)src[i];
}
// x = emb[x_idx]  (row stride 64, bf16 for layer-1 WMMA input)
__global__ void k_gather_x(bf16_t* xbf, const float* emb, const int* xidx, long n) {
  long i = (long)blockIdx.x * blockDim.x + threadIdx.x;
  if (i >= n) return;
  int node = (int)(i >> 6), c = (int)(i & 63);
  xbf[i] = (bf16_t)emb[(long)xidx[node] * 64 + c];
}
// degree + sum of incoming embedded edge attrs (for self-loop fill 'mean')
__global__ void k_deg_loop(float* deg, float* loopsum, const float* emb,
                           const int* eidx, const int* eattr, long n) {
  long i = (long)blockIdx.x * blockDim.x + threadIdx.x;
  if (i >= n) return;
  int e = (int)(i >> 6), c = (int)(i & 63);
  int d = eidx[NEDGES + e];
  atomicAdd(&loopsum[(long)d * 64 + c], emb[(long)eattr[e] * 64 + c]);
  if (c == 0) atomicAdd(&deg[d], 1.0f);
}
// ea2 (bf16): first E rows = emb[eattr], last N rows = loop mean
__global__ void k_build_ea2(bf16_t* ea2, const float* emb, const int* eattr,
                            const float* loopsum, const float* deg, long n) {
  long i = (long)blockIdx.x * blockDim.x + threadIdx.x;
  if (i >= n) return;
  int e = (int)(i >> 6), c = (int)(i & 63);
  float v;
  if (e < NEDGES) v = emb[(long)eattr[e] * 64 + c];
  else            v = loopsum[(long)(e - NEDGES) * 64 + c] / fmaxf(deg[e - NEDGES], 1.0f);
  ea2[i] = (bf16_t)v;
}

// ----------------------- WMMA: xl = x Wl^T + bl ; xr = x Wr^T + br ----------
__global__ __launch_bounds__(32)
void k_lin2_wmma(const bf16_t* __restrict__ xbf,
                 const bf16_t* __restrict__ Wlbf, const bf16_t* __restrict__ Wrbf,
                 const float* __restrict__ bl, const float* __restrict__ br,
                 float* __restrict__ xl, float* __restrict__ xr,
                 int cin, int cout) {
  const int mb = blockIdx.x * 16;          // node tile
  const int cb = blockIdx.y * 16;          // out-channel tile
  const int lane = threadIdx.x;
  const int g = lane >> 4, n16 = lane & 15;

  const bf16_t* arow = xbf + (long)(mb + n16) * cin;
  const bf16_t* blrow = Wlbf + (long)(cb + n16) * cin;  // B lane = column N
  const bf16_t* brrow = Wrbf + (long)(cb + n16) * cin;

  v8f cl = {}, cr = {};
  for (int kk = 0; kk < cin; kk += 32) {
    v16bf a  = load_a_frag(arow, g, kk);
    v16bf bL = *(const v16bf*)(blrow + kk + g * 16);   // 32B aligned
    v16bf bR = *(const v16bf*)(brrow + kk + g * 16);
    cl = wmma_bf16(a, bL, cl);
    cr = wmma_bf16(a, bR, cr);
  }
  const float blv = bl[cb + n16], brv = br[cb + n16];
#pragma unroll
  for (int r = 0; r < 8; ++r) {
    long row = mb + r + 8 * g;
    xl[row * cout + cb + n16] = cl[r] + blv;
    xr[row * cout + cb + n16] = cr[r] + brv;
  }
}

// -------- WMMA fused edge kernel: s[e] = att . lrelu(ea2 We^T + xl[src]+xr[dst])
__global__ __launch_bounds__(32)
void k_score_wmma(const bf16_t* __restrict__ ea2, const bf16_t* __restrict__ Webf,
                  const float* __restrict__ xl, const float* __restrict__ xr,
                  const float* __restrict__ att, const int* __restrict__ eidx,
                  float* __restrict__ s, unsigned* __restrict__ smaxm, int cout) {
  const int eb = blockIdx.x * 16;
  const int lane = threadIdx.x;
  const int g = lane >> 4, n16 = lane & 15;

  const bf16_t* arow = ea2 + (long)(eb + n16) * 64;
  v16bf a0 = load_a_frag(arow, g, 0);     // K = 0..31
  v16bf a1 = load_a_frag(arow, g, 32);    // K = 32..63

  int srcI[8], dstI[8];
#pragma unroll
  for (int r = 0; r < 8; ++r) {
    int e = eb + r + 8 * g;
    srcI[r] = src_of(eidx, e);
    dstI[r] = dst_of(eidx, e);
  }

  float sacc[8];
#pragma unroll
  for (int r = 0; r < 8; ++r) sacc[r] = 0.0f;

  for (int cb = 0; cb < cout; cb += 16) {
    const bf16_t* brow = Webf + (long)(cb + n16) * 64;
    v16bf b0 = *(const v16bf*)(brow + g * 16);
    v16bf b1 = *(const v16bf*)(brow + 32 + g * 16);
    v8f c = {};
    c = wmma_bf16(a0, b0, c);
    c = wmma_bf16(a1, b1, c);
    const float attv = att[cb + n16];
#pragma unroll
    for (int r = 0; r < 8; ++r) {
      float m = c[r] + xl[(long)srcI[r] * cout + cb + n16]
                     + xr[(long)dstI[r] * cout + cb + n16];
      m = (m > 0.0f) ? m : 0.2f * m;      // leaky_relu(0.2)
      sacc[r] += m * attv;
    }
  }
  // row-sum across the 16 lanes of each half-wave group
#pragma unroll
  for (int r = 0; r < 8; ++r) {
    float v = sacc[r];
    v += __shfl_xor(v, 8, 16);
    v += __shfl_xor(v, 4, 16);
    v += __shfl_xor(v, 2, 16);
    v += __shfl_xor(v, 1, 16);
    if (n16 == r) {
      int e = eb + r + 8 * g;
      s[e] = v;
      atomicMax(&smaxm[dstI[r]], fmap(v));
    }
  }
}

// ex = exp(s - smax[dst]); den[dst] += ex
__global__ void k_expden(float* ex, float* den, const float* s,
                         const unsigned* smaxm, const int* eidx) {
  long i = (long)blockIdx.x * blockDim.x + threadIdx.x;
  if (i >= NE2) return;
  int d = dst_of(eidx, (int)i);
  float v = expf(s[i] - funmap(smaxm[d]));
  ex[i] = v;
  atomicAdd(&den[d], v);
}

// in-place: ex[e] <- alpha[e] = ex[e] / den[dst]
__global__ void k_alpha(float* ex, const float* den, const int* eidx) {
  long i = (long)blockIdx.x * blockDim.x + threadIdx.x;
  if (i >= NE2) return;
  int d = dst_of(eidx, (int)i);
  ex[i] = ex[i] / den[d];
}

// acc[dst] += alpha[e] * xl[src], 4 channels per thread (float4)
__global__ void k_aggr4(float* acc, const float* alpha, const float* xl,
                        const int* eidx, int cq /*cout/4*/, long n /*NE2*cq*/) {
  long i = (long)blockIdx.x * blockDim.x + threadIdx.x;
  if (i >= n) return;
  int e = (int)(i / cq);
  int c4 = (int)(i - (long)e * cq) * 4;
  int d = dst_of(eidx, e), sN = src_of(eidx, e);
  float a = alpha[e];
  int cout = cq * 4;
  float4 v = *(const float4*)(xl + (long)sN * cout + c4);
  float* o = acc + (long)d * cout + c4;
  atomicAdd(o + 0, a * v.x);
  atomicAdd(o + 1, a * v.y);
  atomicAdd(o + 2, a * v.z);
  atomicAdd(o + 3, a * v.w);
}

// acc += bias ; optional ELU ; optional bf16 re-emit for next layer's WMMA
__global__ void k_final(float* acc, bf16_t* xbf, const float* bias,
                        int cout, int doElu, int writeBf, long n) {
  long i = (long)blockIdx.x * blockDim.x + threadIdx.x;
  if (i >= n) return;
  int c = (int)(i % cout);
  float v = acc[i] + bias[c];
  if (doElu) v = (v > 0.0f) ? v : expm1f(v);
  acc[i] = v;
  if (writeBf) xbf[i] = (bf16_t)v;
}

// mean pooling: binary-search graph id in ptr, accumulate 4 ch/thread
__global__ void k_pool4(float* out, const float* acc, const int* ptr,
                        long n /*NNODES*64*/) {
  long i = (long)blockIdx.x * blockDim.x + threadIdx.x;
  if (i >= n) return;
  int node = (int)(i >> 6), c4 = (int)(i & 63) * 4;
  int lo = 0, hi = NGRAPH;
  while (lo + 1 < hi) { int mid = (lo + hi) >> 1; if (ptr[mid] <= node) lo = mid; else hi = mid; }
  float4 v = *(const float4*)(acc + (long)node * 256 + c4);
  float* o = out + (long)lo * 256 + c4;
  atomicAdd(o + 0, v.x);
  atomicAdd(o + 1, v.y);
  atomicAdd(o + 2, v.z);
  atomicAdd(o + 3, v.w);
}
__global__ void k_pooldiv(float* out, const int* ptr, long n) {
  long i = (long)blockIdx.x * blockDim.x + threadIdx.x;
  if (i >= n) return;
  int gid = (int)(i >> 8);
  out[i] /= fmaxf((float)(ptr[gid + 1] - ptr[gid]), 1.0f);
}

// ---------------------------------------------------------------------------
static inline int nblk(long n) { return (int)((n + 255) / 256); }

struct Ws {
  bf16_t *xbf, *ea2bf, *wlbf, *wrbf, *webf;
  float *xl, *xr, *acc, *s, *ex, *den, *deg, *loopsum;
  unsigned* smaxm;
};

static void run_layer(hipStream_t st, const Ws& w, const int* eidx,
                      const float* Wl, const float* bl, const float* Wr,
                      const float* br, const float* We, const float* att,
                      const float* bias, int cin, int cout, int doElu, int writeBf) {
  const int TB = 256;
  k_f2bf<<<nblk((long)cout * cin), TB, 0, st>>>(w.wlbf, Wl, (long)cout * cin);
  k_f2bf<<<nblk((long)cout * cin), TB, 0, st>>>(w.wrbf, Wr, (long)cout * cin);
  k_f2bf<<<nblk((long)cout * 64), TB, 0, st>>>(w.webf, We, (long)cout * 64);

  dim3 ggemm(NNODES / 16, cout / 16);
  k_lin2_wmma<<<ggemm, 32, 0, st>>>(w.xbf, w.wlbf, w.wrbf, bl, br, w.xl, w.xr,
                                    cin, cout);

  k_init_u32<<<nblk(NNODES), TB, 0, st>>>(w.smaxm, NNODES, 0x007FFFFFu); // fmap(-inf)
  k_zero_f32<<<nblk(NNODES), TB, 0, st>>>(w.den, NNODES);
  k_zero_f32<<<nblk((long)NNODES * cout), TB, 0, st>>>(w.acc, (long)NNODES * cout);

  k_score_wmma<<<NE2 / 16, 32, 0, st>>>(w.ea2bf, w.webf, w.xl, w.xr, att, eidx,
                                        w.s, w.smaxm, cout);
  k_expden<<<nblk(NE2), TB, 0, st>>>(w.ex, w.den, w.s, w.smaxm, eidx);
  k_alpha<<<nblk(NE2), TB, 0, st>>>(w.ex, w.den, eidx);
  const int cq = cout / 4;
  k_aggr4<<<nblk((long)NE2 * cq), TB, 0, st>>>(w.acc, w.ex, w.xl, eidx, cq,
                                               (long)NE2 * cq);
  k_final<<<nblk((long)NNODES * cout), TB, 0, st>>>(w.acc, w.xbf, bias, cout,
                                                    doElu, writeBf,
                                                    (long)NNODES * cout);
}

extern "C" void kernel_launch(void* const* d_in, const int* in_sizes, int n_in,
                              void* d_out, int out_size, void* d_ws, size_t ws_size,
                              hipStream_t stream) {
  (void)in_sizes; (void)n_in; (void)out_size; (void)ws_size;
  const int*   x_idx = (const int*)d_in[0];
  const int*   eidx  = (const int*)d_in[1];   // [2][E]
  const int*   eattr = (const int*)d_in[2];
  const int*   ptr   = (const int*)d_in[3];
  const float* emb   = (const float*)d_in[4];
  // conv1..conv3: Wl, bl, Wr, br, We, att, bias
  const float* c1[7]; const float* c2[7]; const float* c3[7];
  for (int i = 0; i < 7; ++i) { c1[i] = (const float*)d_in[5 + i];
                                c2[i] = (const float*)d_in[12 + i];
                                c3[i] = (const float*)d_in[19 + i]; }

  // workspace carve (256B aligned)
  char* base = (char*)d_ws;
  size_t off = 0;
  auto take = [&](size_t bytes) -> char* {
    char* p = base + off;
    off += (bytes + 255) & ~(size_t)255;
    return p;
  };
  Ws w;
  w.xbf     = (bf16_t*)take((size_t)NNODES * 128 * sizeof(bf16_t));
  w.ea2bf   = (bf16_t*)take((size_t)NE2 * 64 * sizeof(bf16_t));
  w.xl      = (float*)take((size_t)NNODES * 256 * sizeof(float));
  w.xr      = (float*)take((size_t)NNODES * 256 * sizeof(float));
  w.acc     = (float*)take((size_t)NNODES * 256 * sizeof(float));
  w.s       = (float*)take((size_t)NE2 * sizeof(float));
  w.ex      = (float*)take((size_t)NE2 * sizeof(float));
  w.smaxm   = (unsigned*)take((size_t)NNODES * sizeof(unsigned));
  w.den     = (float*)take((size_t)NNODES * sizeof(float));
  w.deg     = (float*)take((size_t)NNODES * sizeof(float));
  w.loopsum = (float*)take((size_t)NNODES * 64 * sizeof(float));
  w.wlbf    = (bf16_t*)take((size_t)256 * 128 * sizeof(bf16_t));
  w.wrbf    = (bf16_t*)take((size_t)256 * 128 * sizeof(bf16_t));
  w.webf    = (bf16_t*)take((size_t)256 * 64 * sizeof(bf16_t));

  const int TB = 256;
  k_zero_f32<<<nblk((long)NGRAPH * 256), TB, 0, stream>>>((float*)d_out,
                                                          (long)NGRAPH * 256);
  k_zero_f32<<<nblk(NNODES), TB, 0, stream>>>(w.deg, NNODES);
  k_zero_f32<<<nblk((long)NNODES * 64), TB, 0, stream>>>(w.loopsum,
                                                         (long)NNODES * 64);
  k_gather_x<<<nblk((long)NNODES * 64), TB, 0, stream>>>(w.xbf, emb, x_idx,
                                                         (long)NNODES * 64);
  k_deg_loop<<<nblk((long)NEDGES * 64), TB, 0, stream>>>(w.deg, w.loopsum, emb,
                                                         eidx, eattr,
                                                         (long)NEDGES * 64);
  k_build_ea2<<<nblk((long)NE2 * 64), TB, 0, stream>>>(w.ea2bf, emb, eattr,
                                                       w.loopsum, w.deg,
                                                       (long)NE2 * 64);

  run_layer(stream, w, eidx, c1[0], c1[1], c1[2], c1[3], c1[4], c1[5], c1[6],
            64, 128, /*elu=*/1, /*writeBf=*/1);
  run_layer(stream, w, eidx, c2[0], c2[1], c2[2], c2[3], c2[4], c2[5], c2[6],
            128, 128, /*elu=*/1, /*writeBf=*/1);
  run_layer(stream, w, eidx, c3[0], c3[1], c3[2], c3[3], c3[4], c3[5], c3[6],
            128, 256, /*elu=*/0, /*writeBf=*/0);

  k_pool4<<<nblk((long)NNODES * 64), TB, 0, stream>>>((float*)d_out, w.acc, ptr,
                                                      (long)NNODES * 64);
  k_pooldiv<<<nblk((long)NGRAPH * 256), TB, 0, stream>>>((float*)d_out, ptr,
                                                         (long)NGRAPH * 256);
}
